// GraphAttentionLayer_773094114148
// MI455X (gfx1250) — matrix operand order repaired and checked by reading, not compiled, not run on hardware
//
#include <hip/hip_runtime.h>
#include <hip/hip_bf16.h>

typedef __attribute__((ext_vector_type(16))) _Float16 v16h;
typedef __attribute__((ext_vector_type(8)))  _Float16 v8h;
typedef __attribute__((ext_vector_type(4)))  _Float16 v4h;
typedef __attribute__((ext_vector_type(8)))  float    v8f;
typedef __attribute__((ext_vector_type(4)))  float    v4f;
typedef __attribute__((ext_vector_type(4)))  int      v4i;

#define N_NODES 6144
#define DIN 128
#define HEADS 4
#define HD 32
#define OUTD 128
static constexpr float L2E = 1.4426950408889634f;   // log2(e)

// ---------------------------------------------------------------------------
// Kernel 1: h = x @ W[head]  (f32 compute), emit:
//   h16T[head][d][n]  (f16, transposed for WMMA B-fragment contiguity)
//   s_dst[head][n] = h . a[:HD],  s_src[head][n] = h . a[HD:]
// ---------------------------------------------------------------------------
__global__ __launch_bounds__(256) void gat_feat_kernel(
    const float* __restrict__ x,      // [N][128]
    const float* __restrict__ W,      // [H][128][32]
    const float* __restrict__ a,      // [H][64][1]
    _Float16*    __restrict__ h16T,   // [H][32][N]
    float*       __restrict__ s_dst,  // [H][N]
    float*       __restrict__ s_src)  // [H][N]
{
    const int head = blockIdx.y;
    const int n0   = blockIdx.x * 16;
    const int tid  = threadIdx.x;

    __shared__ float xs[16][DIN];     // 8 KB
    __shared__ float ws[DIN][HD];     // 16 KB

    for (int i = tid; i < 16 * DIN; i += 256) {
        int r = i >> 7, c = i & 127;
        xs[r][c] = x[(size_t)(n0 + r) * DIN + c];
    }
    for (int i = tid; i < DIN * HD; i += 256) {
        ws[i >> 5][i & 31] = W[(size_t)head * DIN * HD + i];
    }
    __syncthreads();

    const int d  = tid & 31;          // lane = feature index
    const int wv = tid >> 5;          // wave = local node index
    const float adst = a[head * 2 * HD + d];
    const float asrc = a[head * 2 * HD + HD + d];

    #pragma unroll
    for (int half = 0; half < 2; ++half) {
        const int r = wv + half * 8;
        float acc = 0.f;
        #pragma unroll 8
        for (int k = 0; k < DIN; ++k)
            acc = fmaf(xs[r][k], ws[k][d], acc);

        h16T[(size_t)(head * HD + d) * N_NODES + n0 + r] = (_Float16)acc;

        float pd = acc * adst;
        float ps = acc * asrc;
        #pragma unroll
        for (int off = 16; off > 0; off >>= 1) {
            pd += __shfl_xor(pd, off, 32);
            ps += __shfl_xor(ps, off, 32);
        }
        if (d == 0) {
            s_dst[head * N_NODES + n0 + r] = pd;
            s_src[head * N_NODES + n0 + r] = ps;
        }
    }
}

// ---------------------------------------------------------------------------
// Kernel 1b: maxdst[head] = max_j s_dst[head][j]  (tiny: 4 blocks)
// ---------------------------------------------------------------------------
__global__ __launch_bounds__(256) void gat_maxdst_kernel(
    const float* __restrict__ s_dst, float* __restrict__ maxdst)
{
    const int head = blockIdx.x;
    const int tid  = threadIdx.x;
    __shared__ float red[8];

    float m = -INFINITY;
    for (int j = tid; j < N_NODES; j += 256)
        m = fmaxf(m, s_dst[head * N_NODES + j]);
    #pragma unroll
    for (int off = 16; off > 0; off >>= 1)
        m = fmaxf(m, __shfl_xor(m, off, 32));
    if ((tid & 31) == 0) red[tid >> 5] = m;
    __syncthreads();
    if (tid == 0) {
        float r = red[0];
        #pragma unroll
        for (int w = 1; w < 8; ++w) r = fmaxf(r, red[w]);
        maxdst[head] = r;
    }
}

// ---------------------------------------------------------------------------
// Kernel 2: SINGLE-PASS fused masked-softmax + (attention @ h).
// Shift M_i = lrelu(s_src[i] + max_j s_dst) >= e[i,j] (lrelu monotone), so
// p~ = exp(e - M_i) <= 1 (f16-safe); the shift cancels on division by the
// row sum, which is computed by a 3rd WMMA against an all-ones B matrix so
// numerator and denominator see identical f16 quantization.
// Per-element math in exp2 domain:
//   (lrelu(q+s)-m)*L2E = max((q-m)*L2E + s*L2E, (0.2q-m)*L2E + 0.2*s*L2E)
// grid = (HEADS, N/16): the 4 blocks sharing adj rows are adjacent (L2 reuse).
// ---------------------------------------------------------------------------
__global__ __launch_bounds__(256) void gat_attn_kernel(
    const int*      __restrict__ adj,    // [N][N]
    const _Float16* __restrict__ h16T,   // [H][32][N]
    const float*    __restrict__ s_dst,  // [H][N]
    const float*    __restrict__ s_src,  // [H][N]
    const float*    __restrict__ maxdst, // [H]
    float*          __restrict__ out)    // [N][128]
{
    const int head = blockIdx.x;
    const int i0   = blockIdx.y * 16;
    const int tid  = threadIdx.x;
    const int wave = tid >> 5;
    const int lane = tid & 31;

    __shared__ float sQm[16];                          // (q-m)*L2E per row
    __shared__ float sQ2[16];                          // (0.2q-m)*L2E per row
    __shared__ __align__(32) _Float16 pT[8][16 * 32];  // 8 KB per-wave A tiles
    __shared__ float cRed[8][512];                     // 16 KB per-wave C tiles
    __shared__ float sRed[8][16];                      // per-wave row sums

    if (tid < 16) {
        const float ss = s_src[head * N_NODES + i0 + tid];
        const float v  = ss + maxdst[head];
        const float m  = fmaxf(v, 0.2f * v);           // lrelu upper bound
        sQm[tid] = (ss - m) * L2E;
        sQ2[tid] = (0.2f * ss - m) * L2E;
    }
    __syncthreads();

    const int cg = lane & 7;     // column group: cols cg*4 .. cg*4+3
    const int rg = lane >> 3;    // row group:    rows rg*4 .. rg*4+3

    // Pin this lane's 4 row-constants in registers.
    float qm[4], q2[4];
    #pragma unroll
    for (int i = 0; i < 4; ++i) {
        qm[i] = sQm[rg * 4 + i];
        q2[i] = sQ2[rg * 4 + i];
    }

    const float* sdst = s_dst + (size_t)head * N_NODES;
    const _Float16* hb = h16T + (size_t)(head * HD) * N_NODES;

    const int M    = lane & 15;
    const int kofs = (lane >> 4) * 8;    // unused; A fragment also uses klo
    const int klo  = (lane >> 4) * 16;   // B-fragment K half: 0 or 16

    v16h bOnes;
    #pragma unroll
    for (int t = 0; t < 16; ++t) bOnes[t] = (_Float16)1.0f;

    v8f c0 = {};        // d columns 0..15 (unnormalized)
    v8f c1 = {};        // d columns 16..31
    v8f c2 = {};        // row sums (P @ ones)

    for (int jb = wave * 32; jb < N_NODES; jb += 8 * 32) {
        const int col0 = jb + cg * 4;
        const v4f sd4 = *(const v4f*)(sdst + col0);
        float sA[4], sB[4];
        #pragma unroll
        for (int c = 0; c < 4; ++c) {
            sA[c] = sd4[c] * L2E;
            sB[c] = sA[c] * 0.2f;
        }
        #pragma unroll
        for (int i = 0; i < 4; ++i) {
            const int row = i0 + rg * 4 + i;
            const v4i ad = *(const v4i*)(adj + (size_t)row * N_NODES + col0);
            v4h pp;
            #pragma unroll
            for (int c = 0; c < 4; ++c) {
                const float arg = fmaxf(qm[i] + sA[c], q2[i] + sB[c]);
                float p = __builtin_amdgcn_exp2f(arg); // native v_exp_f32, <= 1
                p = (ad[c] > 0) ? p : 0.f;             // == exp(-9e15 - M)
                pp[c] = (_Float16)p;
            }
            *(v4h*)(&pT[wave][(rg * 4 + i) * 32 + cg * 4]) = pp;
        }
        __builtin_amdgcn_wave_barrier();   // order intra-wave LDS store->load

        // A fragment: lanes 0-15 row M, K {0..15}; lanes 16-31 K {16..31}
        const _Float16* ap = &pT[wave][M * 32];
        v8h alo = *(const v8h*)(ap + klo);
        v8h ahi = *(const v8h*)(ap + klo + 8);
        v16h aFrag = __builtin_shufflevector(alo, ahi,
            0, 1, 2, 3, 4, 5, 6, 7, 8, 9, 10, 11, 12, 13, 14, 15);

        // B fragments from transposed h16T: row = d column, contiguous K halves
        v16h b0 = *(const v16h*)(hb + (size_t)(M)      * N_NODES + jb + klo);
        v16h b1 = *(const v16h*)(hb + (size_t)(16 + M) * N_NODES + jb + klo);

        c0 = __builtin_amdgcn_wmma_f32_16x16x32_f16(false, aFrag, false, b0,
                                                    (short)0, c0, false, false);
        c1 = __builtin_amdgcn_wmma_f32_16x16x32_f16(false, aFrag, false, b1,
                                                    (short)0, c1, false, false);
        c2 = __builtin_amdgcn_wmma_f32_16x16x32_f16(false, aFrag, false, bOnes,
                                                    (short)0, c2, false, false);
    }

    // Row sums live in every column of c2; column N=0 is lanes 0 and 16.
    if ((lane & 15) == 0) {
        const int mh = (lane >> 4) * 8;    // M = v + 8*(lane>=16)
        #pragma unroll
        for (int v = 0; v < 8; ++v) sRed[wave][mh + v] = c2[v];
    }

    // Publish per-wave C tiles: VGPR v -> M = v + 8*(lane>=16), N = lane&15.
    {
        const int Nc = lane & 15;
        const int mh = (lane >> 4) * 8;
        #pragma unroll
        for (int v = 0; v < 8; ++v) {
            cRed[wave][0 * 256 + (mh + v) * 16 + Nc] = c0[v];
            cRed[wave][1 * 256 + (mh + v) * 16 + Nc] = c1[v];
        }
    }
    __syncthreads();

    // Cross-wave reduce + normalize + store.
    for (int idx = tid; idx < 512; idx += 256) {
        const int t   = idx >> 8;
        const int rem = idx & 255;
        const int Mr  = rem >> 4;
        const int Nc  = rem & 15;
        float acc = 0.f;
        float rs  = 0.f;
        #pragma unroll
        for (int w = 0; w < 8; ++w) {
            acc += cRed[w][idx];
            rs  += sRed[w][Mr];
        }
        out[(size_t)(i0 + Mr) * OUTD + head * HD + t * 16 + Nc] = acc / rs;
    }
}

// ---------------------------------------------------------------------------
extern "C" void kernel_launch(void* const* d_in, const int* in_sizes, int n_in,
                              void* d_out, int out_size, void* d_ws, size_t ws_size,
                              hipStream_t stream) {
    const float* x   = (const float*)d_in[0];
    const int*   adj = (const int*)d_in[1];
    const float* W   = (const float*)d_in[2];
    const float* a   = (const float*)d_in[3];
    float* out = (float*)d_out;

    char* ws = (char*)d_ws;
    size_t off = 0;
    auto carve = [&](size_t bytes) -> void* {
        void* p = ws + off;
        off += (bytes + 255) & ~(size_t)255;
        return p;
    };
    _Float16* h16T = (_Float16*)carve((size_t)HEADS * HD * N_NODES * sizeof(_Float16)); // 1.5 MB
    float*    sdst = (float*)carve((size_t)HEADS * N_NODES * sizeof(float));
    float*    ssrc = (float*)carve((size_t)HEADS * N_NODES * sizeof(float));
    float*    mdst = (float*)carve((size_t)HEADS * sizeof(float));

    dim3 g1(N_NODES / 16, HEADS);
    gat_feat_kernel<<<g1, dim3(256), 0, stream>>>(x, W, a, h16T, sdst, ssrc);

    gat_maxdst_kernel<<<dim3(HEADS), dim3(256), 0, stream>>>(sdst, mdst);

    dim3 g2(HEADS, N_NODES / 16);   // head fastest -> adj row reuse in L2
    gat_attn_kernel<<<g2, dim3(256), 0, stream>>>(adj, h16T, sdst, ssrc, mdst, out);
}
